// textGNN_22376779612651
// MI455X (gfx1250) — matrix operand order repaired and checked by reading, not compiled, not run on hardware
//
#include <hip/hip_runtime.h>
#include <hip/hip_bf16.h>

// ---------------------------------------------------------------------------
// 2-layer GCN on MI455X (gfx1250).
// - bf16 WMMA GEMMs with pre-converted/pre-swizzled operands:
//   inner loop = 4x global_load_b128 + 1x v_wmma_f32_16x16x32_bf16
// - L2-resident edge scatter/gather with global_atomic_add_f32
// ---------------------------------------------------------------------------

#define GCN_N   50000
#define GCN_E   800000
#define GCN_INC 256
#define GCN_HID 128
#define GCN_OUT 64

typedef __attribute__((ext_vector_type(16))) __bf16        v16bf;
typedef __attribute__((ext_vector_type(8)))  float         v8f;
typedef __attribute__((ext_vector_type(4)))  unsigned int  v4u;

union Frag16 { v16bf v; v4u q[2]; unsigned int u[8]; };

__device__ __forceinline__ unsigned int f2bf(float f) {
    unsigned int u = __float_as_uint(f);
    unsigned int r = u + 0x7FFFu + ((u >> 16) & 1u);   // round-to-nearest-even
    return r >> 16;
}
__device__ __forceinline__ unsigned int pack2bf(float lo, float hi) {
    return f2bf(lo) | (f2bf(hi) << 16);
}

// ---------------------------------------------------------------------------
// Conversion / swizzle kernels (run once per call, streaming)
// ---------------------------------------------------------------------------

// out[i] = packed bf16 pair of in[2i], in[2i+1]
__global__ void gcn_cvt_bf16(const float* __restrict__ in,
                             unsigned int* __restrict__ out, long npairs) {
    long i = (long)blockIdx.x * blockDim.x + threadIdx.x;
    if (i < npairs) out[i] = pack2bf(in[2 * i], in[2 * i + 1]);
}

// Pre-swizzle row-major W[K][Ncols] (f32) into per-lane WMMA B-fragment
// layout: dword index = tile*256 + lane*8 + v, tile = (k/32)*(Ncols/16)+(n/16).
// Lane holds col n = col0+(lane&15); k = kbase + (lane>=16?16:0) + 2v (+1).
__global__ void gcn_swizzle_w(const float* __restrict__ W,
                              unsigned int* __restrict__ Wf,
                              int K, int Ncols) {
    int idx = blockIdx.x * blockDim.x + threadIdx.x;
    int total = (K >> 5) * (Ncols >> 4) * 256;
    if (idx >= total) return;
    int v    = idx & 7;
    int lane = (idx >> 3) & 31;
    int tile = idx >> 8;
    int nTiles = Ncols >> 4;
    int col0  = (tile % nTiles) << 4;
    int kbase = (tile / nTiles) << 5;
    int n  = col0 + (lane & 15);
    int k0 = kbase + ((lane >> 4) << 4) + 2 * v;
    Wf[idx] = pack2bf(W[(long)k0 * Ncols + n], W[(long)(k0 + 1) * Ncols + n]);
}

// ---------------------------------------------------------------------------
// GEMM: C[M][NCOLS] (f32) = Abf[M][K] (bf16, row-major) * Wfrag (pre-swizzled)
// blockDim.x = (NCOLS/16)*32 (one wave per 16-col tile), gridDim.x = M/16.
// ---------------------------------------------------------------------------
template <int K, int NCOLS>
__global__ void gcn_gemm_bf16_wmma(const unsigned short* __restrict__ Abf,
                                   const unsigned int* __restrict__ Bfrag,
                                   float* __restrict__ C) {
    int wave = threadIdx.x >> 5;
    int lane = threadIdx.x & 31;
    int row0 = blockIdx.x * 16;
    int col0 = wave * 16;

    // A: lane holds row m = row0+(lane&15); K-halves offset by kk = (lane>=16)*16
    int m  = row0 + (lane & 15);
    int kk = (lane >> 4) * 16;
    const unsigned int* arow = (const unsigned int*)(Abf + (long)m * K);

    v8f acc = {};
#pragma unroll
    for (int kb = 0; kb < K; kb += 32) {
        Frag16 a, b;
        // A fragment: two contiguous 16B dword groups per lane
        a.q[0] = *(const v4u*)(arow + ((kb + kk) >> 1));        // K = kb+kk   .. +7
        a.q[1] = *(const v4u*)(arow + ((kb + 8 + kk) >> 1));    // K = kb+8+kk .. +7
        // B fragment: fully contiguous 32B per lane in swizzled buffer
        const unsigned int* bp =
            Bfrag + ((long)((kb >> 5) * (NCOLS >> 4) + (col0 >> 4)) << 8) + lane * 8;
        b.q[0] = *(const v4u*)(bp);
        b.q[1] = *(const v4u*)(bp + 4);
        acc = __builtin_amdgcn_wmma_f32_16x16x32_bf16(
            false, a.v, false, b.v, (short)0, acc, false, false);
    }

    // D layout: VGPR r -> M = row0 + r + (lane>=16 ? 8 : 0); N = col0 + (lane&15)
    int n  = col0 + (lane & 15);
    int mb = row0 + ((lane >> 4) << 3);
#pragma unroll
    for (int r = 0; r < 8; ++r) {
        C[(long)(mb + r) * NCOLS + n] = acc[r];
    }
}

// ---------------------------------------------------------------------------
// Graph kernels
// ---------------------------------------------------------------------------

__global__ void gcn_fill(float* __restrict__ p, float v, long n) {
    long i = (long)blockIdx.x * blockDim.x + threadIdx.x;
    if (i < n) p[i] = v;
}

// deg[dst] += 1 per edge (deg pre-initialized to 1.0 for the self loop)
__global__ void gcn_degree(const long long* __restrict__ dstI,
                           float* __restrict__ deg, int E) {
    int e = blockIdx.x * blockDim.x + threadIdx.x;
    if (e < E) atomicAdd(&deg[dstI[e]], 1.0f);
}

__global__ void gcn_rsqrt(float* __restrict__ d, int n) {
    int i = blockIdx.x * blockDim.x + threadIdx.x;
    if (i < n) {
        float v = d[i];
        d[i] = (v > 0.0f) ? rsqrtf(v) : 0.0f;
    }
}

// One wave32 per edge, 4 channels/lane: acc[dst][c] += feat[src][c]*norm.
// feat and acc (25.6MB each) are L2-resident -> gather + atomics served by L2.
__global__ void gcn_edge_agg128(const long long* __restrict__ srcI,
                                const long long* __restrict__ dstI,
                                const float* __restrict__ dinv,
                                const float* __restrict__ feat,
                                float* __restrict__ acc, int E) {
    int wave = (int)(((long)blockIdx.x * blockDim.x + threadIdx.x) >> 5);
    int lane = threadIdx.x & 31;
    if (wave >= E) return;                 // wave-uniform
    long long s = srcI[wave];
    long long d = dstI[wave];
    float nrm = dinv[s] * dinv[d];
    const float4 v = *(const float4*)(feat + (long)s * 128 + lane * 4);
    float* ad = acc + (long)d * 128 + lane * 4;
    atomicAdd(&ad[0], v.x * nrm);
    atomicAdd(&ad[1], v.y * nrm);
    atomicAdd(&ad[2], v.z * nrm);
    atomicAdd(&ad[3], v.w * nrm);
}

// One wave32 per edge, 2 channels/lane (64-channel layer)
__global__ void gcn_edge_agg64(const long long* __restrict__ srcI,
                               const long long* __restrict__ dstI,
                               const float* __restrict__ dinv,
                               const float* __restrict__ feat,
                               float* __restrict__ acc, int E) {
    int wave = (int)(((long)blockIdx.x * blockDim.x + threadIdx.x) >> 5);
    int lane = threadIdx.x & 31;
    if (wave >= E) return;
    long long s = srcI[wave];
    long long d = dstI[wave];
    float nrm = dinv[s] * dinv[d];
    const float2 v = *(const float2*)(feat + (long)s * 64 + lane * 2);
    float* ad = acc + (long)d * 64 + lane * 2;
    atomicAdd(&ad[0], v.x * nrm);
    atomicAdd(&ad[1], v.y * nrm);
}

// h = relu(agg + hw*dinv^2 + b), written as packed bf16 (layer-2 A operand).
// One thread per channel pair.
__global__ void gcn_post_relu_bf16(const float* __restrict__ agg,
                                   const float* __restrict__ hw,
                                   const float* __restrict__ dinv,
                                   const float* __restrict__ bias,
                                   unsigned int* __restrict__ hbf,
                                   long npairs) {
    long idx = (long)blockIdx.x * blockDim.x + threadIdx.x;
    if (idx >= npairs) return;
    long base = idx * 2;
    long i = base >> 7;                    // / GCN_HID
    int  c = (int)(base & 127);
    float di = dinv[i];
    float d2 = di * di;
    float v0 = fmaxf(agg[base]     + hw[base]     * d2 + bias[c],     0.0f);
    float v1 = fmaxf(agg[base + 1] + hw[base + 1] * d2 + bias[c + 1], 0.0f);
    hbf[idx] = pack2bf(v0, v1);
}

// out += hw*dinv^2 + b   (final layer, no relu)
__global__ void gcn_post_out(float* __restrict__ out,
                             const float* __restrict__ hw,
                             const float* __restrict__ dinv,
                             const float* __restrict__ bias,
                             long total) {
    long idx = (long)blockIdx.x * blockDim.x + threadIdx.x;
    if (idx >= total) return;
    long i = idx >> 6;                     // / GCN_OUT
    int  c = (int)(idx & 63);
    float di = dinv[i];
    out[idx] = out[idx] + hw[idx] * di * di + bias[c];
}

// ---------------------------------------------------------------------------

extern "C" void kernel_launch(void* const* d_in, const int* in_sizes, int n_in,
                              void* d_out, int out_size, void* d_ws, size_t ws_size,
                              hipStream_t stream) {
    const float*     x    = (const float*)d_in[0];
    const long long* eidx = (const long long*)d_in[1];   // int64 [2, E]
    const float*     W1   = (const float*)d_in[2];
    const float*     b1   = (const float*)d_in[3];
    const float*     W2   = (const float*)d_in[4];
    const float*     b2   = (const float*)d_in[5];
    float* out = (float*)d_out;

    const int N = GCN_N, E = GCN_E, INC = GCN_INC, HID = GCN_HID, OUT = GCN_OUT;
    const long long* srcI = eidx;       // edge_index[0] = sources
    const long long* dstI = eidx + E;   // edge_index[1] = targets

    // Workspace carve-up (all chunks 16B-aligned):
    char* w = (char*)d_ws;
    float* dinv = (float*)w;            w += (size_t)N * 4;             // N f32
    unsigned int* xbf = (unsigned int*)w; w += (size_t)N * INC * 2;     // N*INC bf16
    unsigned int* w1f = (unsigned int*)w; w += (size_t)INC * HID * 2;   // swizzled W1
    unsigned int* w2f = (unsigned int*)w; w += (size_t)HID * OUT * 2;   // swizzled W2
    float* hw1  = (float*)w;            w += (size_t)N * HID * 4;       // x@W1
    float* agg1 = (float*)w;            w += (size_t)N * HID * 4;       // edge sum L1
    unsigned int* hbf = (unsigned int*)w; w += (size_t)N * HID * 2;     // h bf16
    float* hw2  = (float*)w;            w += (size_t)N * OUT * 4;       // h@W2

    const int T = 256;
    // init: deg=1 (self loop), accumulators = 0
    gcn_fill<<<(N + T - 1) / T, T, 0, stream>>>(dinv, 1.0f, N);
    gcn_fill<<<((long)N * HID + T - 1) / T, T, 0, stream>>>(agg1, 0.0f, (long)N * HID);
    gcn_fill<<<((long)N * OUT + T - 1) / T, T, 0, stream>>>(out, 0.0f, (long)N * OUT);

    // operand preparation: x -> bf16, W1/W2 -> swizzled bf16 fragments
    gcn_cvt_bf16<<<((long)N * INC / 2 + T - 1) / T, T, 0, stream>>>(x, xbf,
                                                                    (long)N * INC / 2);
    gcn_swizzle_w<<<(INC * HID / 2 + T - 1) / T, T, 0, stream>>>(W1, w1f, INC, HID);
    gcn_swizzle_w<<<(HID * OUT / 2 + T - 1) / T, T, 0, stream>>>(W2, w2f, HID, OUT);

    // symmetric normalization D^-1/2
    gcn_degree<<<(E + T - 1) / T, T, 0, stream>>>(dstI, dinv, E);
    gcn_rsqrt<<<(N + T - 1) / T, T, 0, stream>>>(dinv, N);

    // layer 1: hw1 = x @ W1 (WMMA), scatter, self-loop + bias + relu -> bf16 h
    gcn_gemm_bf16_wmma<GCN_INC, GCN_HID>
        <<<N / 16, (HID / 16) * 32, 0, stream>>>((const unsigned short*)xbf, w1f, hw1);
    gcn_edge_agg128<<<(E + 7) / 8, T, 0, stream>>>(srcI, dstI, dinv, hw1, agg1, E);
    gcn_post_relu_bf16<<<((long)N * HID / 2 + T - 1) / T, T, 0, stream>>>(
        agg1, hw1, dinv, b1, hbf, (long)N * HID / 2);

    // layer 2: hw2 = h @ W2 (WMMA), scatter into d_out, self-loop + bias
    gcn_gemm_bf16_wmma<GCN_HID, GCN_OUT>
        <<<N / 16, (OUT / 16) * 32, 0, stream>>>((const unsigned short*)hbf, w2f, hw2);
    gcn_edge_agg64<<<(E + 7) / 8, T, 0, stream>>>(srcI, dstI, dinv, hw2, out, E);
    gcn_post_out<<<((long)N * OUT + T - 1) / T, T, 0, stream>>>(out, hw2, dinv, b2,
                                                                (long)N * OUT);
}
